// GlobalAttentionLayer_27376121545201
// MI455X (gfx1250) — matrix operand order repaired and checked by reading, not compiled, not run on hardware
//
#include <hip/hip_runtime.h>
#include <hip/hip_bf16.h>

#define BATCH  64
#define SEQ    2048
#define HIDDEN 512
#define ATT    128
#define LN_EPS 1e-3f

typedef __attribute__((ext_vector_type(16))) __bf16       v16bf;
typedef __attribute__((ext_vector_type(8)))  float        v8f;
typedef __attribute__((ext_vector_type(4)))  unsigned int u32x4;
typedef __attribute__((ext_vector_type(8)))  int          i32x8;
typedef __attribute__((ext_vector_type(4)))  int          i32x4;

// ---------------------------------------------------------------------------
// Kernel 0: one-shot transpose + fp32->bf16 convert of W_a into workspace:
// wT[n][k] = (bf16) W_a[k][n]   (128 x 512 bf16, row stride 512)
// grid = 64, block = 256
// ---------------------------------------------------------------------------
__global__ void __launch_bounds__(256)
convert_w_kernel(const float* __restrict__ W_a, __bf16* __restrict__ wT) {
    int idx = blockIdx.x * 1024 + threadIdx.x;        // 64*1024 = 65536
    #pragma unroll
    for (int i = 0; i < 4; ++i, idx += 256) {
        const int n = idx >> 9;        // 0..127
        const int k = idx & 511;       // 0..511
        wT[(size_t)n * HIDDEN + k] = (__bf16)W_a[(size_t)k * ATT + n];
    }
}

// ---------------------------------------------------------------------------
// Kernel 1: context = mean over L of hs, then u_att = context @ U_a
// grid = BATCH, block = 256
// ---------------------------------------------------------------------------
__global__ void __launch_bounds__(256)
ctx_uatt_kernel(const float* __restrict__ hs,
                const float* __restrict__ U_a,
                float* __restrict__ uatt) {
    const int b   = blockIdx.x;
    const int tid = threadIdx.x;
    __shared__ float ctx[HIDDEN];

    const float* base = hs + (size_t)b * SEQ * HIDDEN;
    const int h0 = tid, h1 = tid + 256;
    float acc0 = 0.f, acc1 = 0.f;
    for (int l = 0; l < SEQ; ++l) {
        const float* row = base + (size_t)l * HIDDEN;
        acc0 += row[h0];
        acc1 += row[h1];
    }
    ctx[h0] = acc0 * (1.0f / SEQ);
    ctx[h1] = acc1 * (1.0f / SEQ);
    __syncthreads();

    if (tid < ATT) {
        float u = 0.f;
        for (int h = 0; h < HIDDEN; ++h)
            u += ctx[h] * U_a[h * ATT + tid];
        uatt[b * ATT + tid] = u;
    }
}

// ---------------------------------------------------------------------------
// Kernel 2: h_att = hs @ W_a via v_wmma_f32_16x16x32_bf16, with the W_a
// K-chunks delivered by the Tensor Data Mover (tensor_load_to_lds), double
// buffered and synchronized with s_wait_tensorcnt + workgroup barrier.
// grid = (SEQ/128, BATCH), block = 256 (8 waves). Wave w owns M-rows
// [128*bx + 16w, +16) x all 128 N columns (8 tiles of 16x16).
// ---------------------------------------------------------------------------
__global__ void __launch_bounds__(256)
hatt_wmma_kernel(const float* __restrict__ hs,
                 const __bf16* __restrict__ wT,
                 float* __restrict__ hatt) {
    const int b      = blockIdx.y;
    const int m0     = blockIdx.x * 128;
    const int tid    = threadIdx.x;
    const int wave   = tid >> 5;
    const int lane   = tid & 31;
    const int laneLo = lane & 15;
    const int laneHi = lane >> 4;

    // double-buffered W_a K-chunk, transposed bf16: wlds[buf][n][k_local]
    __shared__ alignas(128) __bf16 wlds[2][ATT][32];

    // --- TDM descriptor pieces (all wave-uniform) ---
    const unsigned long long gbase = (unsigned long long)wT;  // byte address
    const unsigned lds0 = (unsigned)(unsigned long long)&wlds[0][0][0];
    const unsigned lds1 = (unsigned)(unsigned long long)&wlds[1][0][0];

    auto issue_tdm = [&](unsigned ldsAddr, int k0) {
        const unsigned long long ga = gbase + (unsigned long long)(k0 * 2);
        // D# group 0: count=1 | lds_addr | global_addr[56:0] | type=2
        u32x4 g0 = { 1u,
                     ldsAddr,
                     (unsigned)ga,
                     ((unsigned)(ga >> 32) & 0x01FFFFFFu) | (2u << 30) };
        // D# group 1: data_size=2B; tensor 512(k) x 128(n), stride0=512;
        // tile 32(k) x 128(n)
        i32x8 g1 = { 0x00010000,                 // data_size=1 (2 bytes)
                     (int)(512u << 16),          // tensor_dim0 lo16 @ bits 63:48
                     (int)(128u << 16),          // dim0 hi16 | tensor_dim1 lo16
                     (int)(32u << 16),           // dim1 hi16 | tile_dim0
                     128,                        // tile_dim1 | tile_dim2=0
                     512,                        // tensor_dim0_stride[31:0]
                     0, 0 };
        i32x4 z4 = { 0, 0, 0, 0 };
#if __clang_major__ >= 23
        i32x8 z8 = { 0, 0, 0, 0, 0, 0, 0, 0 };
        __builtin_amdgcn_tensor_load_to_lds(g0, g1, z4, z4, z8, 0);
#else
        __builtin_amdgcn_tensor_load_to_lds(g0, g1, z4, z4, 0);
#endif
    };

    v8f acc[8] = {};

    const int rowM = m0 + wave * 16 + laneLo;
    const float* arow = hs + ((size_t)b * SEQ + rowM) * HIDDEN;

    if (wave == 0) issue_tdm(lds0, 0);           // prologue: chunk 0 -> buf0

    for (int step = 0; step < 16; ++step) {
        const int k0 = step * 32;
        __syncthreads();   // previous compute done before its buffer is reused
        if (wave == 0) {
            if (step + 1 < 16) {
                issue_tdm((step & 1) ? lds0 : lds1, k0 + 32);
                __builtin_amdgcn_s_wait_tensorcnt((short)1); // retire current
            } else {
                __builtin_amdgcn_s_wait_tensorcnt((short)0);
            }
        }
        __syncthreads();   // buffer for this step is visible to all waves

        // A fragment (16x32 bf16): lane<16 -> K = k0+{0..15};
        // lane>=16 -> K = k0+{16..31}
        const float* ap = arow + k0 + laneHi * 16;
        __builtin_prefetch(ap + 32, 0, 1);       // next K-chunk of hs
        float4 a0 = *(const float4*)(ap);
        float4 a1 = *(const float4*)(ap + 4);
        float4 a2 = *(const float4*)(ap + 8);
        float4 a3 = *(const float4*)(ap + 12);
        v16bf af;
        af[0]  = (__bf16)a0.x; af[1]  = (__bf16)a0.y;
        af[2]  = (__bf16)a0.z; af[3]  = (__bf16)a0.w;
        af[4]  = (__bf16)a1.x; af[5]  = (__bf16)a1.y;
        af[6]  = (__bf16)a1.z; af[7]  = (__bf16)a1.w;
        af[8]  = (__bf16)a2.x; af[9]  = (__bf16)a2.y;
        af[10] = (__bf16)a2.z; af[11] = (__bf16)a2.w;
        af[12] = (__bf16)a3.x; af[13] = (__bf16)a3.y;
        af[14] = (__bf16)a3.z; af[15] = (__bf16)a3.w;

        const int cur = step & 1;
        // B fragments: lane<16 -> col=laneLo, K=0..15; lane>=16 -> K=16..31
        #pragma unroll
        for (int n = 0; n < 8; ++n) {
            const v16bf bf =
                *(const v16bf*)&wlds[cur][n * 16 + laneLo][laneHi * 16];
            acc[n] = __builtin_amdgcn_wmma_f32_16x16x32_bf16(
                false, af, false, bf, (short)0, acc[n], false, false);
        }
    }

    // C/D layout: VGPR r -> M = r + 8*laneHi, N = laneLo
    float* orow = hatt + ((size_t)b * SEQ + (m0 + wave * 16)) * ATT;
    #pragma unroll
    for (int n = 0; n < 8; ++n) {
        #pragma unroll
        for (int r = 0; r < 8; ++r) {
            int rr = r + laneHi * 8;
            orow[(size_t)rr * ATT + n * 16 + laneLo] = acc[n][r];
        }
    }
}

// ---------------------------------------------------------------------------
// Kernel 3: scores = v . tanh(h_att + u_att); softmax over L; write attn
// grid = BATCH, block = 256 (each thread owns 8 L positions)
// ---------------------------------------------------------------------------
__global__ void __launch_bounds__(256)
scores_softmax_kernel(const float* __restrict__ hatt,
                      const float* __restrict__ uatt,
                      const float* __restrict__ v_a,
                      float* __restrict__ attn) {
    const int b   = blockIdx.x;
    const int tid = threadIdx.x;
    __shared__ float u[ATT];
    __shared__ float v[ATT];
    __shared__ float red[256];

    if (tid < ATT) {
        u[tid] = uatt[b * ATT + tid];
        v[tid] = v_a[tid];
    }
    __syncthreads();

    float s[8];
    float m = -3.402823466e38f;
    #pragma unroll
    for (int j = 0; j < 8; ++j) {
        const int l = tid + j * 256;
        const float* hp = hatt + ((size_t)b * SEQ + l) * ATT;
        float acc = 0.f;
        for (int a = 0; a < ATT; ++a)
            acc += v[a] * tanhf(hp[a] + u[a]);
        s[j] = acc;
        m = fmaxf(m, acc);
    }

    red[tid] = m; __syncthreads();
    for (int o = 128; o > 0; o >>= 1) {
        if (tid < o) red[tid] = fmaxf(red[tid], red[tid + o]);
        __syncthreads();
    }
    m = red[0]; __syncthreads();

    float sum = 0.f;
    #pragma unroll
    for (int j = 0; j < 8; ++j) { s[j] = __expf(s[j] - m); sum += s[j]; }
    red[tid] = sum; __syncthreads();
    for (int o = 128; o > 0; o >>= 1) {
        if (tid < o) red[tid] += red[tid + o];
        __syncthreads();
    }
    const float inv = 1.0f / red[0];
    #pragma unroll
    for (int j = 0; j < 8; ++j)
        attn[(size_t)b * SEQ + tid + j * 256] = s[j] * inv;
}

// ---------------------------------------------------------------------------
// Kernel 4: attended = attn^T @ hs per batch, then LayerNorm
// grid = BATCH, block = 256 (each thread owns 2 H positions)
// ---------------------------------------------------------------------------
__global__ void __launch_bounds__(256)
attend_ln_kernel(const float* __restrict__ hs,
                 const float* __restrict__ attn,
                 const float* __restrict__ gamma,
                 const float* __restrict__ beta,
                 float* __restrict__ out) {
    const int b   = blockIdx.x;
    const int tid = threadIdx.x;
    __shared__ float aw[SEQ];
    __shared__ float red[256];

    for (int i = tid; i < SEQ; i += 256)
        aw[i] = attn[(size_t)b * SEQ + i];
    __syncthreads();

    const float* base = hs + (size_t)b * SEQ * HIDDEN;
    const int h0 = tid, h1 = tid + 256;
    float acc0 = 0.f, acc1 = 0.f;
    for (int l = 0; l < SEQ; ++l) {
        const float* row = base + (size_t)l * HIDDEN;
        const float w = aw[l];
        acc0 += w * row[h0];
        acc1 += w * row[h1];
    }

    red[tid] = acc0 + acc1; __syncthreads();
    for (int o = 128; o > 0; o >>= 1) {
        if (tid < o) red[tid] += red[tid + o];
        __syncthreads();
    }
    const float mu = red[0] * (1.0f / HIDDEN); __syncthreads();

    const float d0 = acc0 - mu, d1 = acc1 - mu;
    red[tid] = d0 * d0 + d1 * d1; __syncthreads();
    for (int o = 128; o > 0; o >>= 1) {
        if (tid < o) red[tid] += red[tid + o];
        __syncthreads();
    }
    const float var = red[0] * (1.0f / HIDDEN);
    const float rs  = rsqrtf(var + LN_EPS);

    out[(size_t)b * HIDDEN + h0] = d0 * rs * gamma[h0] + beta[h0];
    out[(size_t)b * HIDDEN + h1] = d1 * rs * gamma[h1] + beta[h1];
}

// ---------------------------------------------------------------------------
extern "C" void kernel_launch(void* const* d_in, const int* in_sizes, int n_in,
                              void* d_out, int out_size, void* d_ws, size_t ws_size,
                              hipStream_t stream) {
    (void)in_sizes; (void)n_in; (void)out_size; (void)ws_size;
    const float* hs    = (const float*)d_in[0];
    const float* W_a   = (const float*)d_in[1];
    const float* U_a   = (const float*)d_in[2];
    const float* v_a   = (const float*)d_in[3];
    const float* gamma = (const float*)d_in[4];
    const float* beta  = (const float*)d_in[5];

    float* out  = (float*)d_out;                       // (B, H)
    float* attn = (float*)d_out + BATCH * HIDDEN;      // (B, L)

    float*  hatt = (float*)d_ws;                       // B*L*ATT fp32 (64 MB)
    float*  uatt = hatt + (size_t)BATCH * SEQ * ATT;   // B*ATT
    __bf16* wT   = (__bf16*)(uatt + BATCH * ATT);      // 128x512 bf16 (transposed W_a)

    convert_w_kernel<<<64, 256, 0, stream>>>(W_a, wT);
    ctx_uatt_kernel<<<BATCH, 256, 0, stream>>>(hs, U_a, uatt);
    hatt_wmma_kernel<<<dim3(SEQ / 128, BATCH), 256, 0, stream>>>(hs, wT, hatt);
    scores_softmax_kernel<<<BATCH, 256, 0, stream>>>(hatt, uatt, v_a, attn);
    attend_ln_kernel<<<BATCH, 256, 0, stream>>>(hs, attn, gamma, beta, out);
}
